// Block_53661321396245
// MI455X (gfx1250) — compile-verified, matrix operand order checked by reading, not run on hardware
//
#include <hip/hip_runtime.h>
#include <math.h>

// ---------------------------------------------------------------------------
// Transformer block (pre-LN, causal attention, exact GELU) for MI455X/gfx1250.
// bf16 WMMA (v_wmma_f32_16x16x32_bf16) for every matmul, fp32 accumulate,
// flash-attention online softmax.  GEMM tiles are staged with CDNA5 async
// copies (global_load_async_to_lds_b128 + s_wait_asynccnt), double buffered.
// Weights are pre-transposed to N-major so all async copies are contiguous.
// ---------------------------------------------------------------------------

#define BATCH 4
#define SEQ   2048
#define NH    12
#define HD    64
#define HIDN  768
#define FFN   3072
#define RTOT  (BATCH * SEQ)   // 8192 token rows

typedef __attribute__((ext_vector_type(16))) __bf16       bf16x16;
typedef __attribute__((ext_vector_type(8)))  float        f32x8;
typedef __attribute__((ext_vector_type(4)))  unsigned int u32x4;

union FragU { bf16x16 v; u32x4 q[2]; };

__device__ __forceinline__ unsigned short f32_to_bf16(float f) {
  unsigned int u = __float_as_uint(f);
  u += 0x7fffu + ((u >> 16) & 1u);          // round-to-nearest-even
  return (unsigned short)(u >> 16);
}

// Fragment load for A(16x32)/B(32x16) bf16 layouts: for lane l (g = l>=16),
// row/col = l&15, element e holds K = (e<8 ? e : e+8) + 8*g.  With the tile
// stored K-contiguous this is two 16-byte loads.
__device__ __forceinline__ bf16x16 load_frag(const unsigned short* p) {
  FragU f;
  f.q[0] = *(const u32x4*)(p);        // K = 8g .. 8g+7
  f.q[1] = *(const u32x4*)(p + 16);   // K = 16+8g .. 16+8g+7
  return f.v;
}

__device__ __forceinline__ f32x8 wmma_bf16(bf16x16 a, bf16x16 b, f32x8 c) {
  return __builtin_amdgcn_wmma_f32_16x16x32_bf16(false, a, false, b,
                                                 (short)0, c, false, false);
}

// CDNA5 async copy: 16 bytes per lane, global -> LDS, tracked by ASYNCcnt.
// VDST VGPR carries the LDS byte offset (low 32 bits of the flat LDS address).
__device__ __forceinline__ void async_copy_b128(const void* gsrc, void* ldst) {
  unsigned lofs = (unsigned)(unsigned long long)ldst;
  asm volatile("global_load_async_to_lds_b128 %0, %1, off"
               :: "v"(lofs), "v"(gsrc) : "memory");
}
__device__ __forceinline__ void wait_async_le3() {
  asm volatile("s_wait_asynccnt 3" ::: "memory");
}
__device__ __forceinline__ void wait_async_0() {
  asm volatile("s_wait_asynccnt 0" ::: "memory");
}

// ---------------------------------------------------------------------------
// fp32 (K x N) -> bf16 transposed (N x K), LDS-tiled 32x32.
// Makes GEMM B-tiles contiguous so they can be async-copied.
// ---------------------------------------------------------------------------
__global__ __launch_bounds__(256) void transpose_bf16_kernel(
    const float* __restrict__ in, unsigned short* __restrict__ out,
    int K, int N) {
  __shared__ unsigned short tile[32][33];
  const int k0 = blockIdx.x * 32;
  const int n0 = blockIdx.y * 32;
  const int tx = threadIdx.x;        // 0..31
  const int ty = threadIdx.y;        // 0..7
#pragma unroll
  for (int i = 0; i < 32; i += 8)
    tile[ty + i][tx] = f32_to_bf16(in[(size_t)(k0 + ty + i) * N + n0 + tx]);
  __syncthreads();
#pragma unroll
  for (int i = 0; i < 32; i += 8)
    out[(size_t)(n0 + ty + i) * K + k0 + tx] = tile[tx][ty + i];
}

// ---------------------------------------------------------------------------
// LayerNorm over 768, one 256-thread block per row, bf16 output
// ---------------------------------------------------------------------------
__global__ __launch_bounds__(256) void layernorm_kernel(
    const float* __restrict__ x, const float* __restrict__ w,
    const float* __restrict__ b, unsigned short* __restrict__ out) {
  __shared__ float red[8];
  __shared__ float red2[8];
  const int row = blockIdx.x;
  const int tid = threadIdx.x;
  const float* xr = x + (size_t)row * HIDN;
  float v0 = xr[tid], v1 = xr[tid + 256], v2 = xr[tid + 512];
  float s = v0 + v1 + v2;
#pragma unroll
  for (int m = 1; m < 32; m <<= 1) s += __shfl_xor(s, m, 32);
  if ((tid & 31) == 0) red[tid >> 5] = s;
  __syncthreads();
  float mean = 0.f;
#pragma unroll
  for (int i = 0; i < 8; ++i) mean += red[i];
  mean *= (1.0f / HIDN);
  float d0 = v0 - mean, d1 = v1 - mean, d2 = v2 - mean;
  float s2 = d0 * d0 + d1 * d1 + d2 * d2;
#pragma unroll
  for (int m = 1; m < 32; m <<= 1) s2 += __shfl_xor(s2, m, 32);
  if ((tid & 31) == 0) red2[tid >> 5] = s2;
  __syncthreads();
  float var = 0.f;
#pragma unroll
  for (int i = 0; i < 8; ++i) var += red2[i];
  var *= (1.0f / HIDN);
  float rstd = rsqrtf(var + 1e-5f);
  unsigned short* orow = out + (size_t)row * HIDN;
  orow[tid]       = f32_to_bf16(w[tid]       * d0 * rstd + b[tid]);
  orow[tid + 256] = f32_to_bf16(w[tid + 256] * d1 * rstd + b[tid + 256]);
  orow[tid + 512] = f32_to_bf16(w[tid + 512] * d2 * rstd + b[tid + 512]);
}

// ---------------------------------------------------------------------------
// bf16 GEMM:  out = A(MxK) @ Wt(NxK)^T + bias, fp32 accumulate via WMMA.
// 256 threads = 8 waves; block tile 128x64; wave tile 32x32 (2x2 WMMA tiles).
// Tiles double-buffered in LDS via global_load_async_to_lds_b128.
// MODE 1: split-heads bf16 (B,H,S,D)    MODE 2: exact GELU -> bf16
// MODE 3: fp32 residual add             MODE 4: split-heads transposed (B,H,D,S)
// ---------------------------------------------------------------------------
template <int MODE>
__global__ __launch_bounds__(256) void gemm_bf16(
    const unsigned short* __restrict__ A,   // M x K bf16, row-major
    const unsigned short* __restrict__ Wt,  // N x K bf16 (pre-transposed)
    const float* __restrict__ bias, const float* __restrict__ res,
    void* __restrict__ outv, int M, int K, int N) {
  __shared__ unsigned short As[2][128 * 32];  // [row][k] k-contiguous
  __shared__ unsigned short Bt[2][64 * 32];   // [n][k]   k-contiguous
  const int tid  = threadIdx.x;
  const int lane = tid & 31;
  const int wave = tid >> 5;
  const int g    = lane >> 4;
  const int ln   = lane & 15;
  const int m0   = blockIdx.x * 128;
  const int n0   = blockIdx.y * 64;
  const int wm   = (wave >> 1) * 32;
  const int wn   = (wave & 1) * 32;

  const f32x8 zero = {0.f, 0.f, 0.f, 0.f, 0.f, 0.f, 0.f, 0.f};
  f32x8 acc[2][2];
  acc[0][0] = zero; acc[0][1] = zero; acc[1][0] = zero; acc[1][1] = zero;

  // 3 async b128 copies per thread per k-step: 2 for A (8 KB), 1 for B (4 KB)
  auto issue_tile = [&](int buf, int k0) {
#pragma unroll
    for (int i = 0; i < 2; ++i) {
      int s   = tid + 256 * i;          // 16-byte segment id, A tile
      int row = s >> 2;
      int sg  = (s & 3) * 8;
      async_copy_b128(A + (size_t)(m0 + row) * K + k0 + sg,
                      &As[buf][row * 32 + sg]);
    }
    {
      int row = tid >> 2;               // B tile
      int sg  = (tid & 3) * 8;
      async_copy_b128(Wt + (size_t)(n0 + row) * K + k0 + sg,
                      &Bt[buf][row * 32 + sg]);
    }
  };

  issue_tile(0, 0);
  const int nk = K >> 5;
  for (int t = 0; t < nk; ++t) {
    const int buf = t & 1;
    if (t + 1 < nk) {
      issue_tile(buf ^ 1, (t + 1) << 5);
      wait_async_le3();                 // current buffer's 3 copies complete
    } else {
      wait_async_0();
    }
    __syncthreads();                    // all waves' copies visible
    bf16x16 af[2], bfr[2];
#pragma unroll
    for (int mi = 0; mi < 2; ++mi)
      af[mi] = load_frag(&As[buf][(wm + mi * 16 + ln) * 32 + 8 * g]);
#pragma unroll
    for (int ni = 0; ni < 2; ++ni)
      bfr[ni] = load_frag(&Bt[buf][(wn + ni * 16 + ln) * 32 + 8 * g]);
#pragma unroll
    for (int mi = 0; mi < 2; ++mi)
#pragma unroll
      for (int ni = 0; ni < 2; ++ni)
        acc[mi][ni] = wmma_bf16(af[mi], bfr[ni], acc[mi][ni]);
    __syncthreads();                    // protect buffer reuse (WAR)
  }

  // Epilogue.  C/D layout: col = ln, row = v + 8*g.
#pragma unroll
  for (int mi = 0; mi < 2; ++mi) {
#pragma unroll
    for (int ni = 0; ni < 2; ++ni) {
      int c = n0 + wn + ni * 16 + ln;
      float bia = bias[c];
#pragma unroll
      for (int v = 0; v < 8; ++v) {
        int r = m0 + wm + mi * 16 + v + 8 * g;
        float val = acc[mi][ni][v] + bia;
        if (MODE == 1) {
          int bb = r >> 11, ss = r & (SEQ - 1);
          int hh = c >> 6,  dd = c & (HD - 1);
          ((unsigned short*)outv)[(((size_t)bb * NH + hh) * SEQ + ss) * HD + dd] =
              f32_to_bf16(val);
        } else if (MODE == 2) {
          val = 0.5f * val * (1.0f + erff(val * 0.70710678118f));
          ((unsigned short*)outv)[(size_t)r * N + c] = f32_to_bf16(val);
        } else if (MODE == 3) {
          ((float*)outv)[(size_t)r * N + c] = res[(size_t)r * N + c] + val;
        } else if (MODE == 4) {
          int bb = r >> 11, ss = r & (SEQ - 1);
          int hh = c >> 6,  dd = c & (HD - 1);
          ((unsigned short*)outv)[(((size_t)bb * NH + hh) * HD + dd) * SEQ + ss] =
              f32_to_bf16(val);
        } else {
          ((unsigned short*)outv)[(size_t)r * N + c] = f32_to_bf16(val);
        }
      }
    }
  }
}

// ---------------------------------------------------------------------------
// Flash attention, causal.  One wave = one 16-row q tile; 8 waves/block cover
// 128 q rows of one (b,h).  Q,K in (B,H,S,D): score fragments load straight
// from global (d contiguous).  V in (B,H,D,S): P@V B-fragments also load
// straight from global (kv contiguous).  Only P goes through LDS (transpose
// of the score C-layout into the A-fragment layout).
// ---------------------------------------------------------------------------
__global__ __launch_bounds__(256) void attention_kernel(
    const unsigned short* __restrict__ Qb, const unsigned short* __restrict__ Kb,
    const unsigned short* __restrict__ Vtb, unsigned short* __restrict__ Ob) {
  __shared__ unsigned short Pl[8][16 * 32];  // per-wave P tile [q][kv]
  const int tid  = threadIdx.x;
  const int lane = tid & 31;
  const int wave = tid >> 5;
  const int g    = lane >> 4;
  const int ln   = lane & 15;
  const int bh   = blockIdx.y;
  const int b    = bh / NH;
  const int h    = bh % NH;
  const int q0   = blockIdx.x * 128 + wave * 16;

  const unsigned short* Qp = Qb  + (size_t)bh * SEQ * HD;
  const unsigned short* Kp = Kb  + (size_t)bh * SEQ * HD;
  const unsigned short* Vp = Vtb + (size_t)bh * HD * SEQ;  // [d][s]
  unsigned short* Pw = Pl[wave];

  bf16x16 qf[2];
#pragma unroll
  for (int c = 0; c < 2; ++c)
    qf[c] = load_frag(Qp + (size_t)(q0 + ln) * HD + c * 32 + 8 * g);

  const f32x8 zero = {0.f, 0.f, 0.f, 0.f, 0.f, 0.f, 0.f, 0.f};
  f32x8 o[4] = {zero, zero, zero, zero};
  float mrow[8], lrow[8];
#pragma unroll
  for (int v = 0; v < 8; ++v) { mrow[v] = -3.0e38f; lrow[v] = 0.f; }

  const int ntiles = ((q0 + 15) >> 5) + 1;   // causal kv bound
  for (int t = 0; t < ntiles; ++t) {
    const int kv0 = t * 32;
    // S = Q Kt (two 16-wide column tiles over d = 64)
    f32x8 s0 = zero, s1 = zero;
#pragma unroll
    for (int c = 0; c < 2; ++c) {
      bf16x16 k0f = load_frag(Kp + (size_t)(kv0 + ln) * HD + c * 32 + 8 * g);
      bf16x16 k1f = load_frag(Kp + (size_t)(kv0 + 16 + ln) * HD + c * 32 + 8 * g);
      s0 = wmma_bf16(qf[c], k0f, s0);
      s1 = wmma_bf16(qf[c], k1f, s1);
    }
    // Online softmax; row r lives at (vgpr v = r&7, half g = r>>3).
    float p0a[8], p1a[8], corr[8];
#pragma unroll
    for (int v = 0; v < 8; ++v) {
      const int qr = q0 + v + 8 * g;
      float a0 = (kv0 + ln      <= qr) ? s0[v] * 0.125f : -3.0e38f;
      float a1 = (kv0 + 16 + ln <= qr) ? s1[v] * 0.125f : -3.0e38f;
      float tmax = fmaxf(a0, a1);
#pragma unroll
      for (int m = 1; m < 16; m <<= 1) tmax = fmaxf(tmax, __shfl_xor(tmax, m, 32));
      float mn = fmaxf(mrow[v], tmax);
      float e0 = __expf(a0 - mn);
      float e1 = __expf(a1 - mn);
      float cr = __expf(mrow[v] - mn);
      mrow[v] = mn;
      float ssum = e0 + e1;
#pragma unroll
      for (int m = 1; m < 16; m <<= 1) ssum += __shfl_xor(ssum, m, 32);
      lrow[v] = lrow[v] * cr + ssum;
      corr[v] = cr;
      p0a[v] = e0; p1a[v] = e1;
    }
#pragma unroll
    for (int j2 = 0; j2 < 4; ++j2)
#pragma unroll
      for (int v = 0; v < 8; ++v) o[j2][v] *= corr[v];
    // Transpose P into A-fragment layout through LDS (wave-local ordering).
#pragma unroll
    for (int v = 0; v < 8; ++v) {
      Pw[(v + 8 * g) * 32 + ln]      = f32_to_bf16(p0a[v]);
      Pw[(v + 8 * g) * 32 + 16 + ln] = f32_to_bf16(p1a[v]);
    }
    asm volatile("s_wait_dscnt 0" ::: "memory");
    bf16x16 pf = load_frag(Pw + ln * 32 + 8 * g);
    // O += P @ V : B-fragments (lane = d column, elements = kv) from global
#pragma unroll
    for (int j2 = 0; j2 < 4; ++j2) {
      bf16x16 vf = load_frag(Vp + (size_t)(j2 * 16 + ln) * SEQ + kv0 + 8 * g);
      o[j2] = wmma_bf16(pf, vf, o[j2]);
    }
  }
  // Normalize, merge heads -> (B,S,HID) bf16
#pragma unroll
  for (int v = 0; v < 8; ++v) {
    float inv = 1.0f / lrow[v];
    int s = q0 + v + 8 * g;
    size_t rowbase = ((size_t)b * SEQ + s) * HIDN + h * HD;
#pragma unroll
    for (int j2 = 0; j2 < 4; ++j2)
      Ob[rowbase + j2 * 16 + ln] = f32_to_bf16(o[j2][v] * inv);
  }
}

// ---------------------------------------------------------------------------
extern "C" void kernel_launch(void* const* d_in, const int* in_sizes, int n_in,
                              void* d_out, int out_size, void* d_ws, size_t ws_size,
                              hipStream_t stream) {
  (void)in_sizes; (void)n_in; (void)out_size; (void)ws_size;
  const float* x    = (const float*)d_in[0];
  const float* ln1w = (const float*)d_in[1];
  const float* ln1b = (const float*)d_in[2];
  const float* Wq   = (const float*)d_in[3];
  const float* bq   = (const float*)d_in[4];
  const float* Wk   = (const float*)d_in[5];
  const float* bk   = (const float*)d_in[6];
  const float* Wv   = (const float*)d_in[7];
  const float* bv   = (const float*)d_in[8];
  const float* Wo   = (const float*)d_in[9];
  const float* bo   = (const float*)d_in[10];
  const float* ln2w = (const float*)d_in[11];
  const float* ln2b = (const float*)d_in[12];
  const float* W1   = (const float*)d_in[13];
  const float* b1   = (const float*)d_in[14];
  const float* W2   = (const float*)d_in[15];
  const float* b2   = (const float*)d_in[16];
  float* out = (float*)d_out;

  // Workspace carve-up (~166 MB total)
  char* ws = (char*)d_ws;
  size_t off = 0;
  auto carve = [&](size_t bytes) -> char* {
    char* p = ws + off;
    off += (bytes + 255) & ~(size_t)255;
    return p;
  };
  unsigned short* wq_t = (unsigned short*)carve((size_t)HIDN * HIDN * 2);
  unsigned short* wk_t = (unsigned short*)carve((size_t)HIDN * HIDN * 2);
  unsigned short* wv_t = (unsigned short*)carve((size_t)HIDN * HIDN * 2);
  unsigned short* wo_t = (unsigned short*)carve((size_t)HIDN * HIDN * 2);
  unsigned short* w1_t = (unsigned short*)carve((size_t)HIDN * FFN * 2);
  unsigned short* w2_t = (unsigned short*)carve((size_t)FFN * HIDN * 2);
  unsigned short* hn_b = (unsigned short*)carve((size_t)RTOT * HIDN * 2);
  unsigned short* q_b  = (unsigned short*)carve((size_t)RTOT * HIDN * 2);
  unsigned short* k_b  = (unsigned short*)carve((size_t)RTOT * HIDN * 2);
  unsigned short* v_t  = (unsigned short*)carve((size_t)RTOT * HIDN * 2);
  unsigned short* at_b = (unsigned short*)carve((size_t)RTOT * HIDN * 2);
  unsigned short* h2_b = (unsigned short*)carve((size_t)RTOT * HIDN * 2);
  unsigned short* m1_b = (unsigned short*)carve((size_t)RTOT * FFN * 2);
  float*          x1   = (float*)carve((size_t)RTOT * HIDN * 4);

  // Weight transposes fp32 (KxN) -> bf16 (NxK); all end up L2-resident (14 MB)
  dim3 tb(32, 8);
  transpose_bf16_kernel<<<dim3(HIDN / 32, HIDN / 32), tb, 0, stream>>>(Wq, wq_t, HIDN, HIDN);
  transpose_bf16_kernel<<<dim3(HIDN / 32, HIDN / 32), tb, 0, stream>>>(Wk, wk_t, HIDN, HIDN);
  transpose_bf16_kernel<<<dim3(HIDN / 32, HIDN / 32), tb, 0, stream>>>(Wv, wv_t, HIDN, HIDN);
  transpose_bf16_kernel<<<dim3(HIDN / 32, HIDN / 32), tb, 0, stream>>>(Wo, wo_t, HIDN, HIDN);
  transpose_bf16_kernel<<<dim3(HIDN / 32, FFN / 32),  tb, 0, stream>>>(W1, w1_t, HIDN, FFN);
  transpose_bf16_kernel<<<dim3(FFN / 32, HIDN / 32),  tb, 0, stream>>>(W2, w2_t, FFN, HIDN);

  // LN1
  layernorm_kernel<<<RTOT, 256, 0, stream>>>(x, ln1w, ln1b, hn_b);

  // QKV projections: Q,K -> (B,H,S,D); V -> (B,H,D,S) for direct PV fragments
  dim3 gHID(RTOT / 128, HIDN / 64);
  gemm_bf16<1><<<gHID, 256, 0, stream>>>(hn_b, wq_t, bq, nullptr, q_b, RTOT, HIDN, HIDN);
  gemm_bf16<1><<<gHID, 256, 0, stream>>>(hn_b, wk_t, bk, nullptr, k_b, RTOT, HIDN, HIDN);
  gemm_bf16<4><<<gHID, 256, 0, stream>>>(hn_b, wv_t, bv, nullptr, v_t, RTOT, HIDN, HIDN);

  // Causal flash attention, merged-head bf16 out
  attention_kernel<<<dim3(SEQ / 128, BATCH * NH), 256, 0, stream>>>(q_b, k_b, v_t, at_b);

  // Output projection + residual: x1 = x + at @ Wo + bo
  gemm_bf16<3><<<gHID, 256, 0, stream>>>(at_b, wo_t, bo, x, x1, RTOT, HIDN, HIDN);

  // LN2
  layernorm_kernel<<<RTOT, 256, 0, stream>>>(x1, ln2w, ln2b, h2_b);

  // MLP: m1 = gelu(h2 @ W1 + b1);  out = x1 + m1 @ W2 + b2
  dim3 gFF(RTOT / 128, FFN / 64);
  gemm_bf16<2><<<gFF, 256, 0, stream>>>(h2_b, w1_t, b1, nullptr, m1_b, RTOT, HIDN, FFN);
  gemm_bf16<3><<<gHID, 256, 0, stream>>>(m1_b, w2_t, b2, x1, out, RTOT, FFN, HIDN);
}
